// DripHead_69509750719110
// MI455X (gfx1250) — compile-verified, matrix operand order checked by reading, not run on hardware
//
#include <hip/hip_runtime.h>
#include <hip/hip_bf16.h>

// ---------------- problem constants ----------------
#define D_TOTAL 1024
#define D_COMP  512
#define BATCH   8
#define TSEQ    2048
#define K2      2048      // 2*D_TOTAL (GEMM K dimension)
#define NWG     64        // persistent workgroups
#define COLS    16        // output columns per WG per gate (64*16 = 1024)
#define TPB     256
#define NWAVE   8
#define KTILES  (K2/32)   // 64 K-tiles of 32 for wmma 16x16x32

typedef __attribute__((ext_vector_type(16))) __bf16 v16bf;
typedef __attribute__((ext_vector_type(8)))  __bf16 v8bf;
typedef __attribute__((ext_vector_type(8)))  float  v8f;
typedef int v4i_vs __attribute__((vector_size(16)));  // matches builtin proto

// ---------------- LDS layout (bytes) ----------------
// xh    : 16 x 2048 bf16 A-operand tile (rows 8..15 zeroed once) = 65536
// wsw[3]: 3 x (2048 x 16) bf16 weight slices, swizzled           = 196608
// acc   : 3 x 256 f32 wmma cross-wave accumulators               = 3072
// u_l   : 256 f32 (update gate, phase A -> phase B)              = 1024
// h_own : 128 f32 (this WG's fp32 h columns)                     = 512
// bias  : 3 x 16 f32 (padded)                                    = 256
#define LDS_XH    0
#define LDS_W     (65536)
#define LDS_ACC   (LDS_W + 196608)
#define LDS_U     (LDS_ACC + 3072)
#define LDS_HOWN  (LDS_U + 1024)
#define LDS_BIAS  (LDS_HOWN + 512)
#define LDS_TOTAL (LDS_BIAS + 256)

// ---------------- async global->LDS copy (CDNA5 path) ----------------
#if defined(__has_builtin)
#if __has_builtin(__builtin_amdgcn_global_load_async_to_lds_b128)
#define HAVE_ASYNC_LDS 1
#endif
#endif

__device__ inline void copy16_g2l(const __bf16* g, __bf16* l) {
#ifdef HAVE_ASYNC_LDS
  // per-lane 16B DMA into LDS, tracked by ASYNCcnt
  __builtin_amdgcn_global_load_async_to_lds_b128(
      (__attribute__((address_space(1))) v4i_vs*)(uintptr_t)g,
      (__attribute__((address_space(3))) v4i_vs*)(uintptr_t)l, 0, 0);
#else
  *(uint4*)l = *(const uint4*)g;
#endif
}

__device__ inline void wait_async_copies() {
#ifdef HAVE_ASYNC_LDS
  asm volatile("s_wait_asynccnt 0x0" ::: "memory");
#endif
}

// ---------------- grid-wide barrier (L2 atomics) ----------------
__device__ inline void grid_barrier(unsigned* cnt, unsigned* gen) {
  __syncthreads();
  if (threadIdx.x == 0) {
    unsigned g = __hip_atomic_load(gen, __ATOMIC_RELAXED, __HIP_MEMORY_SCOPE_AGENT);
    __threadfence();  // release our stores to agent scope
    unsigned prev = __hip_atomic_fetch_add(cnt, 1u, __ATOMIC_ACQ_REL,
                                           __HIP_MEMORY_SCOPE_AGENT);
    if (prev == (unsigned)(gridDim.x - 1)) {
      __hip_atomic_store(cnt, 0u, __ATOMIC_RELAXED, __HIP_MEMORY_SCOPE_AGENT);
      __hip_atomic_fetch_add(gen, 1u, __ATOMIC_RELEASE, __HIP_MEMORY_SCOPE_AGENT);
    } else {
      while (__hip_atomic_load(gen, __ATOMIC_ACQUIRE,
                               __HIP_MEMORY_SCOPE_AGENT) == g) {
        __builtin_amdgcn_s_sleep(1);
      }
    }
  }
  __syncthreads();
  __threadfence();  // acquire: invalidate WGP cache before re-reading h / r*h
}

// ---------------- WMMA operand loaders ----------------
// A bf16 16x32 layout: lane (m = lane&15, half = lane>>4), element e:
//   K = (e/8)*16 + half*8 + (e%8)  -> two contiguous 16B runs per lane.
// Rows 8..15 of xh are zero, so the load is unconditional (EXEC stays full,
// no per-iteration zero materialization).
__device__ inline v16bf load_a_tile(const __bf16* xh, int kt, int lane) {
  int m = lane & 15;
  int lh = lane >> 4;
  const __bf16* base = xh + m * K2 + kt * 32 + lh * 8;
  v8bf lo = *(const v8bf*)(base);
  v8bf hi = *(const v8bf*)(base + 16);
  v16bf a;
#pragma unroll
  for (int i = 0; i < 8; ++i) { a[i] = lo[i]; a[i + 8] = hi[i]; }
  return a;
}

// B tiles pre-swizzled in LDS: lane's 16 elements are 32 contiguous bytes.
__device__ inline v16bf load_b_tile(const __bf16* wsw, int kt, int lane) {
  const v8bf* q = (const v8bf*)(wsw + kt * 512 + lane * 16);
  v8bf lo = q[0];
  v8bf hi = q[1];
  v16bf b;
#pragma unroll
  for (int i = 0; i < 8; ++i) { b[i] = lo[i]; b[i + 8] = hi[i]; }
  return b;
}

__device__ inline void scatter_acc(float* acc, v8f c, int lane) {
  int mb = (lane >> 4) * 8;  // C: vgpr i, lanes<16 -> M=i ; lanes>=16 -> M=i+8
  int n = lane & 15;
#pragma unroll
  for (int i = 0; i < 8; ++i) atomicAdd(&acc[(mb + i) * 16 + n], c[i]);
}

// Fused two-gate GEMM: one A load feeds both W_r and W_u tiles.
__device__ inline void gemm_two(const __bf16* xh, const __bf16* w0,
                                const __bf16* w1, float* acc0, float* acc1,
                                int wave, int lane) {
  v8f c0 = {}, c1 = {};
#pragma unroll
  for (int i = 0; i < KTILES / NWAVE; ++i) {
    int kt = wave + i * NWAVE;
    v16bf a = load_a_tile(xh, kt, lane);
    v16bf b0 = load_b_tile(w0, kt, lane);
    v16bf b1 = load_b_tile(w1, kt, lane);
    c0 = __builtin_amdgcn_wmma_f32_16x16x32_bf16(false, a, false, b0, (short)0,
                                                 c0, false, false);
    c1 = __builtin_amdgcn_wmma_f32_16x16x32_bf16(false, a, false, b1, (short)0,
                                                 c1, false, false);
  }
  scatter_acc(acc0, c0, lane);
  scatter_acc(acc1, c1, lane);
}

__device__ inline void gemm_one(const __bf16* xh, const __bf16* wsw, float* acc,
                                int wave, int lane) {
  v8f c = {};
#pragma unroll
  for (int i = 0; i < KTILES / NWAVE; ++i) {
    int kt = wave + i * NWAVE;
    v16bf a = load_a_tile(xh, kt, lane);
    v16bf b = load_b_tile(wsw, kt, lane);
    c = __builtin_amdgcn_wmma_f32_16x16x32_bf16(false, a, false, b, (short)0, c,
                                                false, false);
  }
  scatter_acc(acc, c, lane);
}

// Convert + swizzle one gate's weight column slice into LDS (B-operand order).
__device__ inline void stage_weights(const float* W, __bf16* wsw, int c0,
                                     int tid) {
  for (int idx = tid; idx < K2 * COLS; idx += TPB) {
    int k = idx >> 4;
    int n = idx & 15;
    float w = W[(size_t)k * D_TOTAL + c0 + n];
    int kt = k >> 5;
    int kin = k & 31;
    int grp = kin >> 4;  // element group (e/8)
    int rem = kin & 15;
    int lh = rem >> 3;   // lane half
    int e = grp * 8 + (rem & 7);
    int l = lh * 16 + n;
    wsw[kt * 512 + l * 16 + e] = (__bf16)w;
  }
}

// ---------------- persistent GRU kernel ----------------
__global__ void __launch_bounds__(TPB, 1) gru_persistent(
    const __bf16* __restrict__ xs,  // [T][8][1024] bf16, precomputed
    const float* __restrict__ W_r, const float* __restrict__ b_r,
    const float* __restrict__ W_u, const float* __restrict__ b_u,
    const float* __restrict__ W_c, const float* __restrict__ b_c,
    __bf16* __restrict__ h_glob, __bf16* __restrict__ rh_glob, unsigned* cnt,
    unsigned* gen, float* __restrict__ out) {
  extern __shared__ float4 smem_raw[];
  char* smem = (char*)smem_raw;
  __bf16* xh = (__bf16*)(smem + LDS_XH);
  __bf16* wswR = (__bf16*)(smem + LDS_W);
  __bf16* wswU = wswR + K2 * COLS;
  __bf16* wswC = wswU + K2 * COLS;
  float* accR = (float*)(smem + LDS_ACC);
  float* accU = accR + 256;
  float* accC = accU + 256;
  float* u_l = (float*)(smem + LDS_U);
  float* h_own = (float*)(smem + LDS_HOWN);  // [8][16] fp32
  float* bias = (float*)(smem + LDS_BIAS);   // [3][16]

  const int tid = threadIdx.x;
  const int lane = tid & 31;
  const int wave = tid >> 5;
  const int c0 = (int)blockIdx.x * COLS;

  // one-time init: weights (bf16, swizzled), bias, zero A rows 8..15, h_own
  stage_weights(W_r, wswR, c0, tid);
  stage_weights(W_u, wswU, c0, tid);
  stage_weights(W_c, wswC, c0, tid);
  if (tid < 16) {
    bias[tid] = b_r[c0 + tid];
    bias[16 + tid] = b_u[c0 + tid];
    bias[32 + tid] = b_c[c0 + tid];
  }
  for (int idx = tid; idx < BATCH * K2; idx += TPB)
    xh[BATCH * K2 + idx] = (__bf16)0.0f;  // rows 8..15 = 0, never rewritten
  if (tid < BATCH * COLS) h_own[tid] = 0.0f;
  __syncthreads();

  const size_t outHalf = (size_t)BATCH * TSEQ * D_COMP;

  for (int t = 0; t < TSEQ; ++t) {
    // ---- phase A: async-stage x_t | h into A tile (pure 16B copies) ----
    {
      const __bf16* xrow = xs + (size_t)t * (BATCH * D_TOTAL);
#pragma unroll
      for (int i = 0; i < 2048 / TPB; ++i) {       // 2048 16B chunks
        int cch = tid + i * TPB;
        int half = cch >> 10;                      // 0: x, 1: h
        int cc = cch & 1023;
        int b = cc >> 7;                           // row
        int o = (cc & 127) * 8;                    // bf16 offset within row
        const __bf16* g = half ? (h_glob + b * D_TOTAL + o)
                               : (xrow + b * D_TOTAL + o);
        copy16_g2l(g, xh + b * K2 + half * D_TOTAL + o);
      }
    }
    {  // init accumulators with bias while copies are in flight
      int m = tid >> 4, n = tid & 15;
      accR[tid] = (m < BATCH) ? bias[n] : 0.0f;
      accU[tid] = (m < BATCH) ? bias[16 + n] : 0.0f;
    }
    wait_async_copies();
    __syncthreads();

    gemm_two(xh, wswR, wswU, accR, accU, wave, lane);
    __syncthreads();

    {
      int m = tid >> 4, n = tid & 15;
      if (m < BATCH) {
        float r = 1.0f / (1.0f + __expf(-accR[tid]));
        float u = 1.0f / (1.0f + __expf(-accU[tid]));
        u_l[tid] = u;
        float hold = h_own[tid];  // fp32 h for this WG's own columns
        rh_glob[m * D_TOTAL + c0 + n] = (__bf16)(r * hold);
      }
    }
    grid_barrier(cnt, gen);  // all r*h columns visible everywhere

    // ---- phase B: candidate c from (x, r*h), then h update ----
#pragma unroll
    for (int i = 0; i < 1024 / TPB; ++i) {
      int cc = tid + i * TPB;
      int b = cc >> 7;
      int o = (cc & 127) * 8;
      copy16_g2l(rh_glob + b * D_TOTAL + o, xh + b * K2 + D_TOTAL + o);
    }
    {
      int m = tid >> 4, n = tid & 15;
      accC[tid] = (m < BATCH) ? bias[32 + n] : 0.0f;
    }
    wait_async_copies();
    __syncthreads();

    gemm_one(xh, wswC, accC, wave, lane);
    __syncthreads();

    {
      int m = tid >> 4, n = tid & 15;
      if (m < BATCH) {
        float cval = tanhf(accC[tid]);
        float u = u_l[tid];
        int col = c0 + n;
        float hold = h_own[tid];
        float hn = (1.0f - u) * hold + u * cval;
        h_own[tid] = hn;
        h_glob[m * D_TOTAL + col] = (__bf16)hn;  // bf16 copy for next GEMM
        size_t row = (size_t)m * TSEQ + t;
        if (col < D_COMP) out[row * D_COMP + col] = hn;
        else out[outHalf + row * D_COMP + (col - D_COMP)] = hn;
      }
    }
    grid_barrier(cnt, gen);  // h fully updated before next step
  }
}

// ---------------- embedding gather -> bf16 xs[T][8][1024] (parallel) -------
__global__ void __launch_bounds__(TPB) embed_kernel(
    const int* __restrict__ tok_ids, const float* __restrict__ emb_r,
    const float* __restrict__ emb_i, __bf16* __restrict__ xs) {
  size_t i = (size_t)blockIdx.x * TPB + threadIdx.x;
  if (i >= (size_t)TSEQ * BATCH * D_TOTAL) return;
  int j = (int)(i & 1023);
  size_t bt = i >> 10;
  int b = (int)(bt & 7);
  int t = (int)(bt >> 3);
  int tk = tok_ids[b * TSEQ + t];
  float v = (j < D_COMP) ? emb_r[(size_t)tk * D_COMP + j]
                         : emb_i[(size_t)tk * D_COMP + (j - D_COMP)];
  xs[i] = (__bf16)v;
}

// ---------------- final LayerNorm (in place over d_out) ----------------
__global__ void __launch_bounds__(TPB) ln_kernel(float* __restrict__ out,
                                                 const float* __restrict__ sr,
                                                 const float* __restrict__ br,
                                                 const float* __restrict__ si,
                                                 const float* __restrict__ bi) {
  __shared__ float red[TPB];
  __shared__ float red2[TPB];
  int row = blockIdx.x;  // 0 .. 2*B*T-1 ; first half = real part
  int half = (row >= BATCH * TSEQ) ? 1 : 0;
  float* p = out + (size_t)row * D_COMP;
  int tid = threadIdx.x;
  float v0 = p[tid];
  float v1 = p[tid + 256];
  red[tid] = v0 + v1;
  red2[tid] = v0 * v0 + v1 * v1;
  __syncthreads();
  for (int s = 128; s > 0; s >>= 1) {
    if (tid < s) {
      red[tid] += red[tid + s];
      red2[tid] += red2[tid + s];
    }
    __syncthreads();
  }
  float mu = red[0] * (1.0f / D_COMP);
  float var = red2[0] * (1.0f / D_COMP) - mu * mu;
  float inv = rsqrtf(var + 1e-6f);
  const float* sc = half ? si : sr;
  const float* bs = half ? bi : br;
  p[tid] = (v0 - mu) * inv * sc[tid] + bs[tid];
  p[tid + 256] = (v1 - mu) * inv * sc[tid + 256] + bs[tid + 256];
}

// ---------------- workspace init (h0 = 0, barrier state = 0) ----------------
__global__ void init_kernel(__bf16* h, __bf16* rh, unsigned* cnt,
                            unsigned* gen) {
  int i = blockIdx.x * blockDim.x + threadIdx.x;
  if (i < BATCH * D_TOTAL) {
    h[i] = (__bf16)0.0f;
    rh[i] = (__bf16)0.0f;
  }
  if (i == 0) {
    *cnt = 0u;
    *gen = 0u;
  }
}

extern "C" void kernel_launch(void* const* d_in, const int* in_sizes, int n_in,
                              void* d_out, int out_size, void* d_ws,
                              size_t ws_size, hipStream_t stream) {
  const int* tok = (const int*)d_in[0];
  const float* emb_r = (const float*)d_in[1];
  const float* emb_i = (const float*)d_in[2];
  const float* W_r = (const float*)d_in[3];
  const float* b_r = (const float*)d_in[4];
  const float* W_u = (const float*)d_in[5];
  const float* b_u = (const float*)d_in[6];
  const float* W_c = (const float*)d_in[7];
  const float* b_c = (const float*)d_in[8];
  const float* ln_rs = (const float*)d_in[9];
  const float* ln_rb = (const float*)d_in[10];
  const float* ln_is = (const float*)d_in[11];
  const float* ln_ib = (const float*)d_in[12];
  float* out = (float*)d_out;

  // workspace: xs bf16 (32 MB, L2-resident) | h bf16 | rh bf16 | barrier
  __bf16* xs = (__bf16*)d_ws;
  __bf16* h_glob = xs + (size_t)TSEQ * BATCH * D_TOTAL;
  __bf16* rh_glob = h_glob + BATCH * D_TOTAL;
  unsigned* cnt = (unsigned*)(rh_glob + BATCH * D_TOTAL);
  unsigned* gen = cnt + 1;

  // allow ~261 KB dynamic LDS (CDNA5 WGP has 320 KB)
  (void)hipFuncSetAttribute((const void*)gru_persistent,
                            hipFuncAttributeMaxDynamicSharedMemorySize,
                            LDS_TOTAL);

  init_kernel<<<32, TPB, 0, stream>>>(h_glob, rh_glob, cnt, gen);
  embed_kernel<<<(int)(((size_t)TSEQ * BATCH * D_TOTAL + TPB - 1) / TPB), TPB,
                 0, stream>>>(tok, emb_r, emb_i, xs);
  gru_persistent<<<NWG, TPB, LDS_TOTAL, stream>>>(
      xs, W_r, b_r, W_u, b_u, W_c, b_c, h_glob, rh_glob, cnt, gen, out);
  ln_kernel<<<2 * BATCH * TSEQ, TPB, 0, stream>>>(out, ln_rs, ln_rb, ln_is,
                                                  ln_ib);
}
